// MSDeformMatchAttn_10136122818976
// MI455X (gfx1250) — matrix operand order repaired and checked by reading, not compiled, not run on hardware
//
#include <hip/hip_runtime.h>
#include <stdint.h>

// MSDeformMatchAttn for MI455X (gfx1250, wave32).
// Hotspot: 2 x (4096 x 4096 x 2304) patch-correlation GEMM = ~155 GFLOP,
// computed as a sum of 9 shifted-feature-map GEMMs (no im2col; sources are
// L2-resident).  v0 staged f32->bf16 conversion inside the GEMM loop (9x
// redundant, visible as heavy v_mov_b16/v_add3 VALU in the disasm).  v1:
//   - one-shot bf16 conversion kernel (ws copies of q/v, 4 MB each)
//   - GEMM staging is now pure b128 global->LDS copy
//   - double-buffered LDS (ping-pong) so global loads overlap the 8-WMMA block
// Workspace: corr 134.2 MB + topvals 0.5 MB + out_pre 8.4 MB + bf16 q/v 8 MB.

#define NBATCH 2
#define LVLS   4
#define HW     1024
#define LEN    4096
#define DM     256
#define NPTS   4
#define PITCH  40           // ushorts per LDS row: 32 data + 8 pad (80B, 16B-aligned)
#define NCHUNK 72           // 9 shifts * (256/32) channel chunks

typedef __attribute__((ext_vector_type(16))) __bf16 bf16x16;
typedef __attribute__((ext_vector_type(8)))  float  f32x8;

struct Frag32B { uint4 lo, hi; };

__device__ __forceinline__ uint32_t pack_bf2(float a, float b) {
  union { float f; uint32_t u; } ua, ub;
  ua.f = a; ub.f = b;
  uint32_t ra = (ua.u + 0x7FFFu + ((ua.u >> 16) & 1u)) >> 16;  // RNE f32->bf16
  uint32_t rb = (ub.u + 0x7FFFu + ((ub.u >> 16) & 1u)) >> 16;
  return (ra & 0xFFFFu) | (rb << 16);
}

// ---------------------------------------------------------------------------
// Kernel 0: one-shot f32 -> packed bf16 (RNE).  8 floats / thread.
// ---------------------------------------------------------------------------
__global__ __launch_bounds__(256) void cvt_bf16(
    const float* __restrict__ src, uint32_t* __restrict__ dst)
{
  const size_t i = ((size_t)blockIdx.x * 256 + threadIdx.x) * 8;
  const float4 a = *(const float4*)(src + i);
  const float4 b = *(const float4*)(src + i + 4);
  *(uint4*)(dst + i / 2) =
      make_uint4(pack_bf2(a.x, a.y), pack_bf2(a.z, a.w),
                 pack_bf2(b.x, b.y), pack_bf2(b.z, b.w));
}

// ---------------------------------------------------------------------------
// Kernel 1: corr[n][q][k] = sum_{s in 3x3} <Q_s[q,:], V_s[k,:]>  (raw dot; the
// 10/d scale folds into the softmax kernel).  128x128 block tile, 8 waves,
// each wave 2(M) x 4(N) wmma_f32_16x16x32_bf16 tiles.  Flattened K loop of 72
// chunks (shift-major) with double-buffered LDS staging.
// ---------------------------------------------------------------------------
__global__ __launch_bounds__(256) void corr_gemm_bf16(
    const unsigned short* __restrict__ qbf, const unsigned short* __restrict__ vbf,
    float* __restrict__ corr)
{
  __shared__ alignas(16) unsigned short As[2][128 * PITCH];
  __shared__ alignas(16) unsigned short Bs[2][128 * PITCH];

  const int n    = blockIdx.z;
  const int M0   = blockIdx.y * 128;
  const int N0   = blockIdx.x * 128;
  const int tid  = threadIdx.x;
  const int lane = tid & 31;
  const int wave = tid >> 5;
  const int wM   = wave >> 1;   // 0..3 -> 32-row strip
  const int wN   = wave & 1;    // 0..1 -> 64-col strip

  f32x8 acc[2][4];
#pragma unroll
  for (int i = 0; i < 2; ++i)
#pragma unroll
    for (int j = 0; j < 4; ++j)
#pragma unroll
      for (int r = 0; r < 8; ++r) acc[i][j][r] = 0.0f;

  const int lr   = tid >> 1;        // loader row in tile: 0..127
  const int half = (tid & 1) * 16;  // loader channel offset: 0 or 16 (bf16 elems)

  const unsigned short* qn = qbf + (size_t)n * LEN * DM;
  const unsigned short* vn = vbf + (size_t)n * LEN * DM;

  // Fragment lane decomposition (ISA 7.12.2, 16-bit A 16x32 / B 32x16):
  const int fm  = lane & 15;
  const int kbA = (lane >> 4) * 8;   // A: K runs {kbA..+7, 16+kbA..+7}
  const int kbB = (lane >> 4) * 8;   // B: K runs {kbB..+7, 16+kbB..+7}

  // Stage chunk c into LDS buffer b.  c = s*8 + (kc/32).
  auto stage = [&](int c, int b) {
    const int s  = c >> 3;
    const int kc = (c & 7) * 32;
    const int dy = s / 3 - 1;
    const int dx = s - (s / 3) * 3 - 1;

    {
      const int rA = M0 + lr;
      const int yA = ((rA & 1023) >> 5) + dy;
      const int xA = (rA & 31) + dx;
      uint4 u0 = {0, 0, 0, 0}, u1 = u0;
      if (((unsigned)yA < 32u) & ((unsigned)xA < 32u)) {
        const uint4* p = (const uint4*)(qn + ((size_t)((rA >> 10) * HW + yA * 32 + xA)) * DM
                                        + kc + half);
        u0 = p[0]; u1 = p[1];
      }
      uint4* dst = (uint4*)&As[b][lr * PITCH + half];
      dst[0] = u0; dst[1] = u1;
    }
    {
      const int rB = N0 + lr;
      const int yB = ((rB & 1023) >> 5) + dy;
      const int xB = (rB & 31) + dx;
      uint4 u0 = {0, 0, 0, 0}, u1 = u0;
      if (((unsigned)yB < 32u) & ((unsigned)xB < 32u)) {
        const uint4* p = (const uint4*)(vn + ((size_t)((rB >> 10) * HW + yB * 32 + xB)) * DM
                                        + kc + half);
        u0 = p[0]; u1 = p[1];
      }
      uint4* dst = (uint4*)&Bs[b][lr * PITCH + half];
      dst[0] = u0; dst[1] = u1;
    }
  };

  stage(0, 0);

  for (int c = 0; c < NCHUNK; ++c) {
    const int cur = c & 1;
    __syncthreads();                 // cur-buffer stores visible; prev reads done
    if (c + 1 < NCHUNK) stage(c + 1, cur ^ 1);

    bf16x16 afrag[2], bfrag[4];
#pragma unroll
    for (int i = 0; i < 2; ++i) {
      const int row = wM * 32 + i * 16 + fm;
      Frag32B t;
      t.lo = *(const uint4*)&As[cur][row * PITCH + kbA];
      t.hi = *(const uint4*)&As[cur][row * PITCH + 16 + kbA];
      afrag[i] = __builtin_bit_cast(bf16x16, t);
    }
#pragma unroll
    for (int j = 0; j < 4; ++j) {
      const int row = wN * 64 + j * 16 + fm;
      Frag32B t;
      t.lo = *(const uint4*)&Bs[cur][row * PITCH + kbB];
      t.hi = *(const uint4*)&Bs[cur][row * PITCH + 16 + kbB];
      bfrag[j] = __builtin_bit_cast(bf16x16, t);
    }
#pragma unroll
    for (int i = 0; i < 2; ++i)
#pragma unroll
      for (int j = 0; j < 4; ++j)
        acc[i][j] = __builtin_amdgcn_wmma_f32_16x16x32_bf16(
            false, afrag[i], false, bfrag[j], (short)0, acc[i][j], false, false);
  }

  float* cbase = corr + (size_t)n * LEN * LEN;
#pragma unroll
  for (int i = 0; i < 2; ++i) {
    const int mb = M0 + wM * 32 + i * 16 + (lane >> 4) * 8;  // VGPR r -> M = mb + r
#pragma unroll
    for (int j = 0; j < 4; ++j) {
      const int nc = N0 + wN * 64 + j * 16 + (lane & 15);
#pragma unroll
      for (int r = 0; r < 8; ++r)
        cbase[(size_t)(mb + r) * LEN + nc] = acc[i][j][r];
    }
  }
}

// ---------------------------------------------------------------------------
// Kernel 2: per (n,q,level) row of 1024: softmax(10*corr/d) denominator + top-4
// (JAX tie-break: smaller index first).  One wave per row; butterfly merges.
// Writes top-4 probs to ws and sampling_locations into d_out.
// ---------------------------------------------------------------------------
__device__ __forceinline__ bool tk_better(float v1, int i1, float v2, int i2) {
  return (v1 > v2) || (v1 == v2 && i1 < i2);
}
__device__ __forceinline__ void tk_insert(float v, int i, float* tv, int* ti) {
  if (tk_better(v, i, tv[3], ti[3])) {
    tv[3] = v; ti[3] = i;
#pragma unroll
    for (int k = 3; k > 0; --k) {
      if (tk_better(tv[k], ti[k], tv[k-1], ti[k-1])) {
        float fv = tv[k]; tv[k] = tv[k-1]; tv[k-1] = fv;
        int   iv = ti[k]; ti[k] = ti[k-1]; ti[k-1] = iv;
      }
    }
  }
}

__global__ __launch_bounds__(256) void softmax_topk(
    const float* __restrict__ corr, float* __restrict__ topvals,
    float* __restrict__ out_loc)
{
  const int tid = threadIdx.x, lane = tid & 31, wave = tid >> 5;
  const int rowid = blockIdx.x * 8 + wave;      // ((n*LEN+q)*4 + t)
  const int t = rowid & 3;
  const int qq = (rowid >> 2) & (LEN - 1);
  const int n = rowid >> 14;
  const float* row = corr + (size_t)(n * LEN + qq) * LEN + (size_t)t * HW;

  const float scale = 10.0f / (float)DM;
  float e[32];
  float tv[4]; int ti[4];
#pragma unroll
  for (int k = 0; k < 4; ++k) { tv[k] = -3.0e38f; ti[k] = 0x7fffffff; }
#pragma unroll
  for (int j = 0; j < 32; ++j) {
    const int idx = j * 32 + lane;
    const float val = row[idx] * scale;
    e[j] = val;
    tk_insert(val, idx, tv, ti);
  }
#pragma unroll
  for (int off = 16; off; off >>= 1) {
    float ov[4]; int oi[4];
#pragma unroll
    for (int k = 0; k < 4; ++k) { ov[k] = __shfl_xor(tv[k], off); oi[k] = __shfl_xor(ti[k], off); }
#pragma unroll
    for (int k = 0; k < 4; ++k) tk_insert(ov[k], oi[k], tv, ti);
  }
  const float maxv = tv[0];
  float ssum = 0.0f;
#pragma unroll
  for (int j = 0; j < 32; ++j) ssum += expf(e[j] - maxv);
#pragma unroll
  for (int off = 16; off; off >>= 1) ssum += __shfl_xor(ssum, off);
  const float inv = 1.0f / ssum;

  if (lane < NPTS) {
    const int p = lane;
    topvals[((size_t)(n * LEN + qq) * 4 + t) * 4 + p] = expf(tv[p] - maxv) * inv;
    const int idx = ti[p];
    const float ow = (float)(idx >> 5);   // floor(idx / h)
    const float oh = (float)(idx & 31);   // idx - ow*h
    const size_t base = (((size_t)(n * LEN + qq) * LVLS + t) * NPTS + p) * 2;
    out_loc[base + 0] = ow * (1.0f / 32.0f);
    out_loc[base + 1] = oh * (1.0f / 32.0f);
  }
}

// ---------------------------------------------------------------------------
// Kernel 3: one wave per (n,q): softmax over 16 top-vals -> attn; bilinear
// sample (grid_sample zeros-padding semantics) and accumulate 256-ch output.
// ---------------------------------------------------------------------------
__global__ __launch_bounds__(256) void sample_attn(
    const float* __restrict__ vflat, const float* __restrict__ topvals,
    const float* __restrict__ loc, float* __restrict__ out_pre)
{
  const int tid = threadIdx.x, lane = tid & 31, wave = tid >> 5;
  const int gq = blockIdx.x * 8 + wave;   // n*LEN + q
  const int n  = gq >> 12;

  const float* tvp = topvals + (size_t)gq * 16;
  float a[16];
  float mx = -3.0e38f;
#pragma unroll
  for (int k = 0; k < 16; ++k) { a[k] = tvp[k]; mx = fmaxf(mx, a[k]); }
  float ssum = 0.0f;
#pragma unroll
  for (int k = 0; k < 16; ++k) { a[k] = expf(a[k] - mx); ssum += a[k]; }
  const float inv = 1.0f / ssum;
#pragma unroll
  for (int k = 0; k < 16; ++k) a[k] *= inv;

  float acc[8];
#pragma unroll
  for (int j = 0; j < 8; ++j) acc[j] = 0.0f;

  const int c0 = lane * 8;
  const float* vn = vflat + (size_t)n * LEN * DM;
  const float* locp = loc + (size_t)gq * 32;

  for (int l = 0; l < LVLS; ++l) {
#pragma unroll
    for (int p = 0; p < NPTS; ++p) {
      const float lx = locp[(l * 4 + p) * 2 + 0];
      const float ly = locp[(l * 4 + p) * 2 + 1];
      const float x = lx * 32.0f - 0.5f;
      const float y = ly * 32.0f - 0.5f;
      const float x0f = floorf(x), y0f = floorf(y);
      const int x0 = (int)x0f, y0 = (int)y0f;
      const float wx1 = x - x0f, wy1 = y - y0f;
      const float aw = a[l * 4 + p];
      const float w[4] = { (1.0f - wy1) * (1.0f - wx1) * aw,
                           (1.0f - wy1) * wx1 * aw,
                           wy1 * (1.0f - wx1) * aw,
                           wy1 * wx1 * aw };
      const int ys[4] = { y0, y0, y0 + 1, y0 + 1 };
      const int xs[4] = { x0, x0 + 1, x0, x0 + 1 };
#pragma unroll
      for (int c = 0; c < 4; ++c) {
        if ((unsigned)xs[c] < 32u && (unsigned)ys[c] < 32u) {
          const float* src = vn + ((size_t)(l * HW + ys[c] * 32 + xs[c])) * DM + c0;
          const float4 v0 = *(const float4*)(src);
          const float4 v1 = *(const float4*)(src + 4);
          const float ww = w[c];
          acc[0] += ww * v0.x; acc[1] += ww * v0.y; acc[2] += ww * v0.z; acc[3] += ww * v0.w;
          acc[4] += ww * v1.x; acc[5] += ww * v1.y; acc[6] += ww * v1.z; acc[7] += ww * v1.w;
        }
      }
    }
  }
  float* dst = out_pre + (size_t)gq * DM + c0;
  *(float4*)(dst)     = make_float4(acc[0], acc[1], acc[2], acc[3]);
  *(float4*)(dst + 4) = make_float4(acc[4], acc[5], acc[6], acc[7]);
}

// ---------------------------------------------------------------------------
// Kernel 4: output projection  out = out_pre @ W_out^T + b_out   (f32 exact).
// One block per q-row; row staged in LDS; W streamed from L2 (256 KB resident).
// ---------------------------------------------------------------------------
__global__ __launch_bounds__(256) void out_proj(
    const float* __restrict__ out_pre, const float* __restrict__ W,
    const float* __restrict__ b, float* __restrict__ outp)
{
  __shared__ float rowbuf[DM];
  const int gq = blockIdx.x;
  rowbuf[threadIdx.x] = out_pre[(size_t)gq * DM + threadIdx.x];
  __syncthreads();
  const float* wr = W + (size_t)threadIdx.x * DM;
  float s = b[threadIdx.x];
#pragma unroll 8
  for (int c = 0; c < DM; ++c) s = fmaf(rowbuf[c], wr[c], s);
  outp[(size_t)gq * DM + threadIdx.x] = s;
}

// ---------------------------------------------------------------------------
extern "C" void kernel_launch(void* const* d_in, const int* in_sizes, int n_in,
                              void* d_out, int out_size, void* d_ws, size_t ws_size,
                              hipStream_t stream) {
  (void)in_sizes; (void)n_in; (void)out_size; (void)ws_size;
  const float* q = (const float*)d_in[0];  // query (2,4096,256)
  const float* v = (const float*)d_in[2];  // input_flatten (2,4096,256)
  const float* W = (const float*)d_in[5];  // W_out (256,256)
  const float* b = (const float*)d_in[6];  // b_out (256)

  float* out = (float*)d_out;                                // (2,4096,256)
  float* loc = out + (size_t)NBATCH * LEN * DM;              // (2,4096,1,4,4,2)

  const size_t ELTS = (size_t)NBATCH * LEN * DM;             // 2,097,152
  float* ws      = (float*)d_ws;
  float* corr    = ws;                                        // 2*4096*4096 f32
  float* topvals = corr + (size_t)NBATCH * LEN * LEN;         // 2*4096*16  f32
  float* out_pre = topvals + (size_t)NBATCH * LEN * 16;       // 2*4096*256 f32
  unsigned short* qbf = (unsigned short*)(out_pre + ELTS);    // bf16 copies
  unsigned short* vbf = qbf + ELTS;

  cvt_bf16<<<dim3((unsigned)(ELTS / 8 / 256)), 256, 0, stream>>>(q, (uint32_t*)qbf);
  cvt_bf16<<<dim3((unsigned)(ELTS / 8 / 256)), 256, 0, stream>>>(v, (uint32_t*)vbf);
  corr_gemm_bf16<<<dim3(32, 32, NBATCH), 256, 0, stream>>>(qbf, vbf, corr);
  softmax_topk <<<dim3(4096),            256, 0, stream>>>(corr, topvals, loc);
  sample_attn  <<<dim3(1024),            256, 0, stream>>>(v, topvals, loc, out_pre);
  out_proj     <<<dim3(NBATCH * LEN),    256, 0, stream>>>(out_pre, W, b, out);
}
